// Aligner_42743514530233
// MI455X (gfx1250) — compile-verified, matrix operand order checked by reading, not run on hardware
//
#include <hip/hip_runtime.h>
#include <hip/hip_bf16.h>

// ---------------------------------------------------------------------------
// Problem constants (from reference setup_inputs)
// ---------------------------------------------------------------------------
#define BATCH 16
#define ZD    256
#define PD    512
#define TYL   2048
#define TXL   512
#define KW5   5
#define NEGV  (-1e9f)
#define INV_TEMP 10.0f

typedef __attribute__((ext_vector_type(16))) __bf16 v16bf;
typedef __attribute__((ext_vector_type(8)))  __bf16 v8bf;
typedef __attribute__((ext_vector_type(8)))  float  v8f;

__device__ inline __bf16 f2bf(float f) { return (__bf16)f; }

__device__ inline v8bf zero_v8bf() {
  v8bf z;
  for (int i = 0; i < 8; ++i) z[i] = f2bf(0.0f);
  return z;
}

// Load one 16x32 bf16 WMMA fragment row from an LDS tile stored [row][32] bf16.
// Per CDNA5 ISA layout: lanes 0-15 hold K = {0..7, 16..23}, lanes 16-31 hold
// K = {8..15, 24..31} -> two 16-byte ds loads per lane.
__device__ inline v16bf load_frag(const __bf16* rowbase, int half) {
  union { v16bf v; v8bf h[2]; } u;
  u.h[0] = *(const v8bf*)(rowbase + 8 * half);
  u.h[1] = *(const v8bf*)(rowbase + 16 + 8 * half);
  return u.v;
}

// Async global -> LDS copy of one 16-byte chunk (CDNA5 async-tensor path,
// tracked by ASYNCcnt; no VGPR round trip).
__device__ inline void async_b128(const void* gptr, void* lptr) {
  unsigned ldsoff = (unsigned)(size_t)lptr;               // low 32b = LDS offset
  unsigned long long ga = (unsigned long long)(size_t)gptr;
  asm volatile("global_load_async_to_lds_b128 %0, %1, off"
               :: "v"(ldsoff), "v"(ga) : "memory");
}
__device__ inline void wait_async0() {
  asm volatile("s_wait_asynccnt 0x0" ::: "memory");
}

// ---------------------------------------------------------------------------
// f32 [B][C][T] -> bf16 [B][T][C] transpose+convert (32x32 LDS tiles)
// ---------------------------------------------------------------------------
__global__ __launch_bounds__(256)
void cvt_transpose_kernel(const float* __restrict__ in, __bf16* __restrict__ out,
                          int C, int T) {
  __shared__ __bf16 tile[32][33];
  const int b = blockIdx.z;
  const int t0 = blockIdx.x * 32, c0 = blockIdx.y * 32;
  const int tl = threadIdx.x & 31, row = threadIdx.x >> 5;  // 8 rows / pass
  const float* ib = in + (size_t)b * C * T;
  __bf16* ob = out + (size_t)b * T * C;
  for (int r = row; r < 32; r += 8)
    tile[r][tl] = f2bf(ib[(size_t)(c0 + r) * T + (t0 + tl)]);  // coalesced in t
  __syncthreads();
  for (int r = row; r < 32; r += 8)
    ob[(size_t)(t0 + r) * C + (c0 + tl)] = tile[tl][r];        // coalesced in c
}

// ---------------------------------------------------------------------------
// Weights f32 [M][Cin][KW] -> bf16 panels [kw][cblk][m][32]
// ---------------------------------------------------------------------------
__global__ __launch_bounds__(256)
void prep_w_kernel(const float* __restrict__ W, __bf16* __restrict__ Wbf,
                   int M, int Cin, int KWc) {
  const long n = (long)M * Cin * KWc;
  const int nCb = Cin >> 5;
  long i = (long)blockIdx.x * blockDim.x + threadIdx.x;
  const long stride = (long)gridDim.x * blockDim.x;
  for (; i < n; i += stride) {
    const int kw = (int)(i % KWc);
    const long mc = i / KWc;
    const int c = (int)(mc % Cin);
    const int m = (int)(mc / Cin);
    const int cb = c >> 5, ci = c & 31;
    Wbf[(((size_t)kw * nCb + cb) * M + m) * 32 + ci] = f2bf(W[i]);
  }
}

// ---------------------------------------------------------------------------
// Conv1d as KW shifted GEMMs (bf16 WMMA, f32 accumulate).
//   Y[b,t,m] = sum_{kw,c} Wp[kw,cb,m,ci] * Xt[b, t+kw-pad, c]  (+bias,epilogue)
// Block: 256 threads = 8 waves, 128(M) x 128(N=t) tile; wave = 64x32 = 4x2 WMMA.
// Staging: async b128 global->LDS copies (2 per thread per tile per K-step).
// mode 0: GELU(exact) -> bf16 [B][T][M].  mode 1: *y_mask -> f32 + bf16 out.
// ---------------------------------------------------------------------------
__global__ __launch_bounds__(256)
void conv_gemm_wmma_kernel(const __bf16* __restrict__ Wp,   // packed panels
                           const float* __restrict__ bias,  // [M]
                           const __bf16* __restrict__ Xt,   // [B][T][Cin]
                           __bf16* __restrict__ Ybf,        // [B][T][M]
                           float*  __restrict__ Yf,         // [B][T][M] (mode 1)
                           const unsigned char* __restrict__ ymask, // [B][T]
                           int M, int Cin, int KWc, int T, int pad, int mode) {
  __shared__ __attribute__((aligned(16))) __bf16 ldsA[128 * 32];
  __shared__ __attribute__((aligned(16))) __bf16 ldsB[128 * 32];

  const int t0 = blockIdx.x * 128;
  const int m0 = blockIdx.y * 128;
  const int b  = blockIdx.z;
  const int tid  = threadIdx.x;
  const int lane = tid & 31;
  const int wave = tid >> 5;
  const int waveM = wave >> 2;   // 0..1  -> 64 rows
  const int waveN = wave & 3;    // 0..3  -> 32 cols
  const int half = lane >> 4;
  const int lr   = lane & 15;
  const int nCb  = Cin >> 5;

  const v8f vzero = {0.f, 0.f, 0.f, 0.f, 0.f, 0.f, 0.f, 0.f};
  v8f acc[4][2];
  for (int mi = 0; mi < 4; ++mi)
    for (int ni = 0; ni < 2; ++ni) acc[mi][ni] = vzero;

  const __bf16* Xb = Xt + (size_t)b * T * Cin;

  for (int kw = 0; kw < KWc; ++kw) {
    const int shift = kw - pad;
    for (int cb = 0; cb < nCb; ++cb) {
      __syncthreads();  // previous tiles fully consumed
      // --- Stage A: contiguous 8 KB weight panel -> LDS (pure async copy)
      const __bf16* Apanel = Wp + (((size_t)kw * nCb + cb) * M + m0) * 32;
      for (int q = tid; q < 512; q += 256)
        async_b128(Apanel + q * 8, &ldsA[q * 8]);
      // --- Stage B: rows t0+n+shift of Xt, 32 channels from cb*32
      for (int q = tid; q < 512; q += 256) {
        const int n = q >> 2, part = q & 3;
        const int t = t0 + n + shift;
        __bf16* dst = &ldsB[n * 32 + part * 8];
        if ((unsigned)t < (unsigned)T)
          async_b128(&Xb[((size_t)t) * Cin + cb * 32 + part * 8], dst);
        else
          *(v8bf*)dst = zero_v8bf();   // zero-padded conv halo
      }
      wait_async0();
      __syncthreads();

      v16bf bfrag[2];
      for (int ni = 0; ni < 2; ++ni)
        bfrag[ni] = load_frag(&ldsB[(waveN * 32 + ni * 16 + lr) * 32], half);
      for (int mi = 0; mi < 4; ++mi) {
        v16bf afrag = load_frag(&ldsA[(waveM * 64 + mi * 16 + lr) * 32], half);
        for (int ni = 0; ni < 2; ++ni)
          acc[mi][ni] = __builtin_amdgcn_wmma_f32_16x16x32_bf16(
              false, afrag, false, bfrag[ni], (short)0, acc[mi][ni], false, false);
      }
    }
  }

  // Epilogue: output in [B][T][M] (token-major) layout
  for (int mi = 0; mi < 4; ++mi)
    for (int ni = 0; ni < 2; ++ni)
      for (int e = 0; e < 8; ++e) {
        const int m = m0 + waveM * 64 + mi * 16 + e + 8 * half;
        const int t = t0 + waveN * 32 + ni * 16 + lr;
        float v = acc[mi][ni][e] + bias[m];
        const size_t off = ((size_t)b * T + t) * M + m;
        if (mode == 0) {
          float g = 0.5f * v * (1.0f + erff(v * 0.70710678118654752f));
          Ybf[off] = f2bf(g);
        } else {
          const float mk = ymask[(size_t)b * T + t] ? 1.0f : 0.0f;
          v *= mk;
          Yf[off]  = v;
          Ybf[off] = f2bf(v);
        }
      }
}

// ---------------------------------------------------------------------------
// invnorm over contiguous rows: A [B][T][C] f32 -> invn[b,t] (wave reduction)
// ---------------------------------------------------------------------------
__global__ __launch_bounds__(256)
void invnorm_rows_kernel(const float* __restrict__ A, float* __restrict__ invn,
                         int C, int T) {
  const int b = blockIdx.y;
  const int t = blockIdx.x * 8 + (threadIdx.x >> 5);
  const int lane = threadIdx.x & 31;
  const float* row = A + ((size_t)b * T + t) * C;
  float s = 0.0f;
  for (int c = lane; c < C; c += 32) {
    const float v = row[c];
    s += v * v;
  }
  for (int o = 16; o > 0; o >>= 1) s += __shfl_down(s, o);
  if (lane == 0) invn[(size_t)b * T + t] = 1.0f / fmaxf(sqrtf(s), 1e-12f);
}

// invnorm over strided columns: A [B][C][T] f32 -> invn[b,t]
__global__ __launch_bounds__(256)
void invnorm_cols_kernel(const float* __restrict__ A, float* __restrict__ invn,
                         int C, int T) {
  __shared__ float red[256];
  const int b = blockIdx.y;
  const int t = blockIdx.x * 64 + (threadIdx.x & 63);
  const int cp = threadIdx.x >> 6;
  const float* Ab = A + (size_t)b * C * T;
  float s = 0.0f;
  for (int c = cp; c < C; c += 4) {
    const float v = Ab[(size_t)c * T + t];
    s += v * v;
  }
  red[threadIdx.x] = s;
  __syncthreads();
  if (cp == 0) {
    const float tot = red[threadIdx.x] + red[threadIdx.x + 64] +
                      red[threadIdx.x + 128] + red[threadIdx.x + 192];
    invn[(size_t)b * T + t] = 1.0f / fmaxf(sqrtf(tot), 1e-12f);
  }
}

// ---------------------------------------------------------------------------
// Similarity GEMM: logp[b,y,x] = dot(zp[b,y,:], mp[b,x,:]) * invz*invm / TEMP
// Both operands token-major bf16 -> pure contiguous async staging.
// ---------------------------------------------------------------------------
__global__ __launch_bounds__(256)
void sim_gemm_wmma_kernel(const __bf16* __restrict__ ZPt,  // [B][TY][C]
                          const __bf16* __restrict__ MPt,  // [B][TX][C]
                          const float* __restrict__ invz,  // [B][TY]
                          const float* __restrict__ invm,  // [B][TX]
                          float* __restrict__ logp,        // [B][TY][TX]
                          int C, int TY, int TX) {
  __shared__ __attribute__((aligned(16))) __bf16 ldsA[128 * 32];
  __shared__ __attribute__((aligned(16))) __bf16 ldsB[128 * 32];

  const int x0 = blockIdx.x * 128;
  const int y0 = blockIdx.y * 128;
  const int b  = blockIdx.z;
  const int tid  = threadIdx.x;
  const int lane = tid & 31;
  const int wave = tid >> 5;
  const int waveM = wave >> 2, waveN = wave & 3;
  const int half = lane >> 4, lr = lane & 15;

  const v8f vzero = {0.f, 0.f, 0.f, 0.f, 0.f, 0.f, 0.f, 0.f};
  v8f acc[4][2];
  for (int mi = 0; mi < 4; ++mi)
    for (int ni = 0; ni < 2; ++ni) acc[mi][ni] = vzero;

  const __bf16* Za = ZPt + ((size_t)b * TY + y0) * C;
  const __bf16* Ma = MPt + ((size_t)b * TX + x0) * C;
  const float* ivz = invz + (size_t)b * TY;
  const float* ivm = invm + (size_t)b * TX;

  for (int c0 = 0; c0 < C; c0 += 32) {
    __syncthreads();
    for (int q = tid; q < 512; q += 256) {
      const int r = q >> 2, part = q & 3;
      async_b128(&Za[(size_t)r * C + c0 + part * 8], &ldsA[r * 32 + part * 8]);
      async_b128(&Ma[(size_t)r * C + c0 + part * 8], &ldsB[r * 32 + part * 8]);
    }
    wait_async0();
    __syncthreads();

    v16bf bfrag[2];
    for (int ni = 0; ni < 2; ++ni)
      bfrag[ni] = load_frag(&ldsB[(waveN * 32 + ni * 16 + lr) * 32], half);
    for (int mi = 0; mi < 4; ++mi) {
      v16bf afrag = load_frag(&ldsA[(waveM * 64 + mi * 16 + lr) * 32], half);
      for (int ni = 0; ni < 2; ++ni)
        acc[mi][ni] = __builtin_amdgcn_wmma_f32_16x16x32_bf16(
            false, afrag, false, bfrag[ni], (short)0, acc[mi][ni], false, false);
    }
  }

  for (int mi = 0; mi < 4; ++mi)
    for (int ni = 0; ni < 2; ++ni)
      for (int e = 0; e < 8; ++e) {
        const int y = y0 + waveM * 64 + mi * 16 + e + 8 * half;
        const int x = x0 + waveN * 32 + ni * 16 + lr;
        logp[((size_t)b * TY + y) * TX + x] =
            acc[mi][ni][e] * ivz[y] * ivm[x] * INV_TEMP;
      }
}

// ---------------------------------------------------------------------------
// Valid lengths from masks
// ---------------------------------------------------------------------------
__global__ __launch_bounds__(256)
void lens_kernel(const unsigned char* __restrict__ ym, const float* __restrict__ xm,
                 int* __restrict__ lens, int TY, int TX) {
  __shared__ int   sy[256];
  __shared__ float sx[256];
  const int b = blockIdx.x, tid = threadIdx.x;
  int   cy = 0;
  float cx = 0.0f;
  for (int i = tid; i < TY; i += 256) cy += ym[(size_t)b * TY + i] ? 1 : 0;
  for (int i = tid; i < TX; i += 256) cx += xm[(size_t)b * TX + i];
  sy[tid] = cy; sx[tid] = cx;
  __syncthreads();
  for (int s = 128; s > 0; s >>= 1) {
    if (tid < s) { sy[tid] += sy[tid + s]; sx[tid] += sx[tid + s]; }
    __syncthreads();
  }
  if (tid == 0) { lens[b * 2] = sy[0]; lens[b * 2 + 1] = (int)(sx[0] + 0.5f); }
}

// ---------------------------------------------------------------------------
// MAS forward DP: one workgroup per batch, 512 lanes over t_x, LDS
// double-buffered row recurrence (one barrier per t_y row).
// ---------------------------------------------------------------------------
__global__ __launch_bounds__(512)
void mas_forward_kernel(const float* __restrict__ logp, float* __restrict__ cum,
                        const int* __restrict__ lens, int TY, int TX) {
  __shared__ float buf[2][TXL];
  const int b = blockIdx.x;
  const int x = threadIdx.x;
  const int txl = lens[b * 2 + 1];
  const float* lp = logp + (size_t)b * TY * TX;
  float* cm = cum + (size_t)b * TY * TX;

  buf[0][x] = (x == 0) ? 0.0f : NEGV;
  __syncthreads();
  int cur = 0;
  for (int y = 0; y < TY; ++y) {
    const float prev    = buf[cur][x];
    const float shifted = (x == 0) ? NEGV : buf[cur][x - 1];
    float v = fmaxf(prev, shifted) + lp[(size_t)y * TX + x];
    if (!((x <= y) && (x < txl))) v = NEGV;
    cm[(size_t)y * TX + x] = v;
    buf[cur ^ 1][x] = v;
    cur ^= 1;
    __syncthreads();
  }
}

// ---------------------------------------------------------------------------
// MAS backtrack + durations + monotone segment starts (scalar per batch)
// ---------------------------------------------------------------------------
__global__ void mas_backtrack_kernel(const float* __restrict__ cum,
                                     const int* __restrict__ lens,
                                     int* __restrict__ targets,  // [B][TY]
                                     int* __restrict__ dur,      // [B][TX]
                                     int* __restrict__ starts,   // [B][TX]
                                     int* __restrict__ dur_out,  // [B][TX] (d_out)
                                     int TY, int TX) {
  const int b = blockIdx.x;
  const float* cm = cum + (size_t)b * TY * TX;
  const int tyl = lens[b * 2], txl = lens[b * 2 + 1];
  for (int x = 0; x < TX; ++x) dur[b * TX + x] = 0;
  int index = 0;
  for (int y = TY - 1; y >= 0; --y) {
    const bool active = (y < tyl);
    if (y == tyl - 1) index = txl - 1;
    targets[b * TY + y] = active ? index : 0;
    if (active) dur[b * TX + index] += 1;
    const int ym1 = (y - 1 > 0) ? (y - 1) : 0;
    const int im1 = (index - 1 > 0) ? (index - 1) : 0;
    const float v_stay = cm[(size_t)ym1 * TX + index];
    const float v_move = cm[(size_t)ym1 * TX + im1];
    const bool move = (index > 0) && ((index == y) || (v_stay < v_move));
    if (active && move) index -= 1;
  }
  int acc = 0;
  for (int x = 0; x < TX; ++x) {
    starts[b * TX + x] = acc;
    const int d = dur[b * TX + x];
    acc += d;
    dur_out[b * TX + x] = d;
  }
}

// ---------------------------------------------------------------------------
// Pooling: monotone path -> contiguous y segment per x; deterministic sums.
// ---------------------------------------------------------------------------
__global__ __launch_bounds__(256)
void pool_kernel(const float* __restrict__ zspec,   // [B][ZD][TY]
                 const int* __restrict__ dur, const int* __restrict__ starts,
                 float* __restrict__ zpool,         // [B][TX][ZD]
                 int ZDc, int TY, int TX) {
  const int x = blockIdx.x, b = blockIdx.y, c = threadIdx.x;
  const int d  = dur[b * TX + x];
  const int s0 = starts[b * TX + x];
  const float* zb = zspec + ((size_t)b * ZDc + c) * TY;
  float s = 0.0f;
  for (int y = s0; y < s0 + d; ++y) s += zb[y];
  zpool[((size_t)b * TX + x) * ZDc + c] = s / fmaxf((float)d, 1.0f);
}

// ---------------------------------------------------------------------------
// Per-row InfoNCE NLL: nll[row] = logsumexp(logits[row,:]) - logits[row,target]
// ---------------------------------------------------------------------------
__global__ __launch_bounds__(128)
void loss_rows_kernel(const float* __restrict__ logp, const int* __restrict__ targets,
                      float* __restrict__ nll, int TX) {
  __shared__ float red[128];
  const int row = blockIdx.x;
  const int tid = threadIdx.x;
  const float* lp = logp + (size_t)row * TX;
  float m = -3.4e38f;
  for (int x = tid; x < TX; x += 128) m = fmaxf(m, lp[x]);
  red[tid] = m;
  __syncthreads();
  for (int s = 64; s > 0; s >>= 1) {
    if (tid < s) red[tid] = fmaxf(red[tid], red[tid + s]);
    __syncthreads();
  }
  m = red[0];
  __syncthreads();
  float e = 0.0f;
  for (int x = tid; x < TX; x += 128) e += expf(lp[x] - m);
  red[tid] = e;
  __syncthreads();
  for (int s = 64; s > 0; s >>= 1) {
    if (tid < s) red[tid] += red[tid + s];
    __syncthreads();
  }
  if (tid == 0) nll[row] = (m + logf(red[0])) - lp[targets[row]];
}

// ---------------------------------------------------------------------------
// Final reduction: masked mean of nll -> loss; text_mask = zeros.
// ---------------------------------------------------------------------------
__global__ __launch_bounds__(256)
void finalize_kernel(const float* __restrict__ nll, const unsigned char* __restrict__ ym,
                     float* __restrict__ out_loss, float* __restrict__ out_tmask,
                     int n, int ntm) {
  __shared__ float rs[256];
  __shared__ float rc[256];
  const int tid = threadIdx.x;
  float s = 0.0f, c = 0.0f;
  for (int i = tid; i < n; i += 256) {
    const float f = ym[i] ? 1.0f : 0.0f;
    s += nll[i] * f;
    c += f;
  }
  rs[tid] = s; rc[tid] = c;
  __syncthreads();
  for (int st = 128; st > 0; st >>= 1) {
    if (tid < st) { rs[tid] += rs[tid + st]; rc[tid] += rc[tid + st]; }
    __syncthreads();
  }
  if (tid == 0) out_loss[0] = rs[0] / fmaxf(rc[0], 1.0f);
  for (int i = tid; i < ntm; i += 256) out_tmask[i] = 0.0f;
}

// ---------------------------------------------------------------------------
// Host-side orchestration
// ---------------------------------------------------------------------------
extern "C" void kernel_launch(void* const* d_in, const int* in_sizes, int n_in,
                              void* d_out, int out_size, void* d_ws, size_t ws_size,
                              hipStream_t stream) {
  const float*         z_spec = (const float*)d_in[0];
  const unsigned char* y_mask = (const unsigned char*)d_in[1];
  const float*         m_p    = (const float*)d_in[2];
  const float*         x_mask = (const float*)d_in[3];
  const float*         W1     = (const float*)d_in[4];
  const float*         b1     = (const float*)d_in[5];
  const float*         W2     = (const float*)d_in[6];
  const float*         b2     = (const float*)d_in[7];
  const float*         W3     = (const float*)d_in[8];
  const float*         b3     = (const float*)d_in[9];

  char* ws = (char*)d_ws;
  size_t off = 0;
  auto alloc = [&](size_t bytes) {
    char* p = ws + off;
    off += (bytes + 255) & ~(size_t)255;
    return p;
  };
  __bf16* X1t    = (__bf16*)alloc((size_t)BATCH * TYL * ZD * 2);  // z_spec^T bf16
  __bf16* MPt    = (__bf16*)alloc((size_t)BATCH * TXL * PD * 2);  // m_p^T bf16
  __bf16* W1bf   = (__bf16*)alloc((size_t)PD * ZD * KW5 * 2);
  __bf16* W2bf   = (__bf16*)alloc((size_t)PD * PD * KW5 * 2);
  __bf16* W3bf   = (__bf16*)alloc((size_t)PD * PD * 2);
  __bf16* h1t    = (__bf16*)alloc((size_t)BATCH * TYL * PD * 2);
  __bf16* h2t    = (__bf16*)alloc((size_t)BATCH * TYL * PD * 2);
  float*  zproj  = (float*) alloc((size_t)BATCH * TYL * PD * 4);  // [B][TY][C]
  __bf16* zprojb = (__bf16*)alloc((size_t)BATCH * TYL * PD * 2);
  float*  logp   = (float*) alloc((size_t)BATCH * TYL * TXL * 4);
  float*  cum    = (float*) alloc((size_t)BATCH * TYL * TXL * 4);
  float*  invz   = (float*) alloc((size_t)BATCH * TYL * 4);
  float*  invm   = (float*) alloc((size_t)BATCH * TXL * 4);
  int*    targets = (int*)  alloc((size_t)BATCH * TYL * 4);
  int*    dur     = (int*)  alloc((size_t)BATCH * TXL * 4);
  int*    starts  = (int*)  alloc((size_t)BATCH * TXL * 4);
  float*  nll     = (float*)alloc((size_t)BATCH * TYL * 4);
  int*    lens    = (int*)  alloc((size_t)BATCH * 2 * 4);

  // d_out: z_pooled [16,512,256] f32 | durations [16,512] i32 | loss | text_mask
  float* out_zpool = (float*)d_out;
  int*   out_dur   = (int*)(out_zpool + (size_t)BATCH * TXL * ZD);
  float* out_loss  = (float*)(out_dur + (size_t)BATCH * TXL);
  float* out_tmask = out_loss + 1;

  // 1. Transpose+convert activations to token-major bf16
  cvt_transpose_kernel<<<dim3(TYL / 32, ZD / 32, BATCH), 256, 0, stream>>>(
      z_spec, X1t, ZD, TYL);
  cvt_transpose_kernel<<<dim3(TXL / 32, PD / 32, BATCH), 256, 0, stream>>>(
      m_p, MPt, PD, TXL);
  // 2. Pack weights into bf16 panels
  prep_w_kernel<<<512, 256, 0, stream>>>(W1, W1bf, PD, ZD, KW5);
  prep_w_kernel<<<512, 256, 0, stream>>>(W2, W2bf, PD, PD, KW5);
  prep_w_kernel<<<512, 256, 0, stream>>>(W3, W3bf, PD, PD, 1);
  // 3. Conv1 + GELU, Conv2 + GELU, 1x1 Conv3 * y_mask
  conv_gemm_wmma_kernel<<<dim3(TYL / 128, PD / 128, BATCH), 256, 0, stream>>>(
      W1bf, b1, X1t, h1t, nullptr, y_mask, PD, ZD, KW5, TYL, KW5 / 2, 0);
  conv_gemm_wmma_kernel<<<dim3(TYL / 128, PD / 128, BATCH), 256, 0, stream>>>(
      W2bf, b2, h1t, h2t, nullptr, y_mask, PD, PD, KW5, TYL, KW5 / 2, 0);
  conv_gemm_wmma_kernel<<<dim3(TYL / 128, PD / 128, BATCH), 256, 0, stream>>>(
      W3bf, b3, h2t, zprojb, zproj, y_mask, PD, PD, 1, TYL, 0, 1);
  // 4. L2 inverse norms
  invnorm_rows_kernel<<<dim3(TYL / 8, BATCH), 256, 0, stream>>>(zproj, invz, PD, TYL);
  invnorm_cols_kernel<<<dim3(TXL / 64, BATCH), 256, 0, stream>>>(m_p, invm, PD, TXL);
  // 5. Cosine similarity / TEMP (logits == log_p)
  sim_gemm_wmma_kernel<<<dim3(TXL / 128, TYL / 128, BATCH), 256, 0, stream>>>(
      zprojb, MPt, invz, invm, logp, PD, TYL, TXL);
  // 6. Mask lengths, MAS DP, backtrack
  lens_kernel<<<BATCH, 256, 0, stream>>>(y_mask, x_mask, lens, TYL, TXL);
  mas_forward_kernel<<<BATCH, TXL, 0, stream>>>(logp, cum, lens, TYL, TXL);
  mas_backtrack_kernel<<<BATCH, 1, 0, stream>>>(cum, lens, targets, dur, starts,
                                                out_dur, TYL, TXL);
  // 7. Pooled z (deterministic contiguous segment sums)
  pool_kernel<<<dim3(TXL, BATCH), ZD, 0, stream>>>(z_spec, dur, starts, out_zpool,
                                                   ZD, TYL, TXL);
  // 8. InfoNCE loss + outputs
  loss_rows_kernel<<<BATCH * TYL, 128, 0, stream>>>(logp, targets, nll, TXL);
  finalize_kernel<<<1, 256, 0, stream>>>(nll, y_mask, out_loss, out_tmask,
                                         BATCH * TYL, BATCH * TXL);
}
